// VNTransmitterUnit_59004260712938
// MI455X (gfx1250) — matrix operand order repaired and checked by reading, not compiled, not run on hardware
//
#include <hip/hip_runtime.h>
#include <hip/hip_bf16.h>
#include <math.h>

// ---------------------------------------------------------------------------
// MI455X (gfx1250, wave32) fused virtual-node attention pipeline.
//   1. cvt f32->f16 of Wq, Wk, W, g           (prep, trivially BW bound)
//   2. zero smax/denom/ctx scratch
//   3. qp  = g16 @ Wq16^T                     (WMMA f32_16x16x32_f16)
//   4. score = tanh(h@Wk^T + qp[cl] + ab)@Ws  (fused WMMA + epilogue, no kp spill)
//      48-row M-blocks to cut Wk16 L2 re-read traffic to ~1 GB
//   5. e = exp(score - smax[cl]); denom[cl] += e
//   6. ctx[cl] += (e/denom[cl]) * h           (f32 atomics into L2-resident 2MB)
//   7. ctx -> f16
//   8. out = tanh(ctx16 @ W16^T + b)          (WMMA + tanh epilogue)
// ---------------------------------------------------------------------------

#define DIM 512
#define LDH (DIM + 8)   // padded LDS row stride in halves (kills bank conflicts)
#define MROWS 48        // rows of h per block in score_kernel (3 M-tiles)

typedef __attribute__((ext_vector_type(16))) _Float16 v16h;
typedef __attribute__((ext_vector_type(8)))  _Float16 half8;
typedef __attribute__((ext_vector_type(8)))  float    v8f;

union ABFrag { v16h v; half8 h[2]; };

// ---------------------------------------------------------------- utilities
__global__ void cvt_f32_f16_kernel(const float* __restrict__ s,
                                   _Float16* __restrict__ d, int n4) {
  int i = blockIdx.x * blockDim.x + threadIdx.x;
  if (i < n4) {
    float4 v = ((const float4*)s)[i];
    union { _Float16 h[4]; unsigned long long u; } p;
    p.h[0] = (_Float16)v.x; p.h[1] = (_Float16)v.y;
    p.h[2] = (_Float16)v.z; p.h[3] = (_Float16)v.w;
    ((unsigned long long*)d)[i] = p.u;
  }
}

__global__ void zero_kernel(unsigned int* __restrict__ p, int n) {
  int i = blockIdx.x * blockDim.x + threadIdx.x;
  if (i < n) p[i] = 0u;
}

// ------------------------------------------------- generic OUT = A @ B^T GEMM
// A16: M x DIM row-major f16.  B16: N x DIM row-major f16 (weight matrix).
// One wave -> one 16x16 tile.  8 waves/block cover 8 N-tiles.
__global__ __launch_bounds__(256) void wmma_gemm_nt_kernel(
    const _Float16* __restrict__ A16, const _Float16* __restrict__ B16,
    const float* __restrict__ bias, float* __restrict__ outp,
    int M, int N, int doTanh) {
  int wave = threadIdx.x >> 5;
  int lane = threadIdx.x & 31;
  int mloc = lane & 15;
  int khalf = lane >> 4;                 // 0: lanes 0-15, 1: lanes 16-31
  int mt = blockIdx.x;
  int nt = blockIdx.y * 8 + wave;
  int arow = mt * 16 + mloc; if (arow >= M) arow = M - 1;   // clamp padding rows
  int ncol = nt * 16 + mloc;
  const _Float16* abase = A16 + (size_t)arow * DIM;
  const _Float16* bbase = B16 + (size_t)ncol * DIM;
  v8f acc = {0.f,0.f,0.f,0.f,0.f,0.f,0.f,0.f};
  for (int k0 = 0; k0 < DIM; k0 += 32) {
    ABFrag a, b;
    const _Float16* ap = abase + k0 + khalf * 8;
    a.h[0] = *(const half8*)(ap);
    a.h[1] = *(const half8*)(ap + 16);
    const _Float16* bp = bbase + k0 + khalf * 8;
    b.h[0] = *(const half8*)(bp);
    b.h[1] = *(const half8*)(bp + 16);
    acc = __builtin_amdgcn_wmma_f32_16x16x32_f16(false, a.v, false, b.v,
                                                 (short)0, acc, false, false);
  }
  for (int r = 0; r < 8; ++r) {
    int row = mt * 16 + khalf * 8 + r;
    if (row < M) {
      float v = acc[r];
      if (doTanh) v = tanhf(v + bias[ncol]);
      outp[(size_t)row * N + ncol] = v;
    }
  }
}

// --------------------------------------------- fused kp -> score per 48 rows
// Block = 256 threads (8 waves). Each block: one 48-row strip of h.
// Per wave: 3 M-tiles x 4 N-tiles (12 accumulators). Per k-step: 3 LDS A-frags
// + 4 global B-frags feed 12 WMMAs, so Wk16 is re-read only once per 48 rows.
// kp never touches memory; epilogue folds qp[cluster] + attn_bias -> tanh ->
// *Ws -> butterfly row reduction -> score + encoded atomicMax segment max.
__global__ __launch_bounds__(256) void score_kernel(
    const float* __restrict__ h, const _Float16* __restrict__ Wk16,
    const float* __restrict__ qp, const int* __restrict__ vn_index,
    const int* __restrict__ n_id, const float* __restrict__ attn_bias,
    const float* __restrict__ Ws, const float* __restrict__ bs,
    float* __restrict__ score, unsigned int* __restrict__ smax_enc, int Bn) {
  __shared__ _Float16 lds_h[MROWS * LDH];      // 48*520*2 = 49,920 B
  __shared__ int      lds_cl[MROWS];
  __shared__ float    lds_partial[8 * MROWS];  // 1.5 KB

  int tid  = threadIdx.x;
  int wave = tid >> 5;
  int lane = tid & 31;
  int row0 = blockIdx.x * MROWS;

  // cooperative load + f32->f16 convert of the 48x512 h strip into LDS
  {
    int lrow0 = tid >> 4;         // 0..15
    int lcol  = tid & 15;         // 0..15
    for (int rr = 0; rr < MROWS / 16; ++rr) {
      int lrow = lrow0 + rr * 16;
      int grow = row0 + lrow;
      bool rv = grow < Bn;
      const float4* src = (const float4*)(h + (size_t)grow * DIM);
      for (int j = 0; j < 8; ++j) {
        int c4 = lcol + j * 16;   // float4 index within the row
        float4 v = rv ? src[c4] : make_float4(0.f, 0.f, 0.f, 0.f);
        union { _Float16 hh[4]; unsigned long long u; } p;
        p.hh[0] = (_Float16)v.x; p.hh[1] = (_Float16)v.y;
        p.hh[2] = (_Float16)v.z; p.hh[3] = (_Float16)v.w;
        *(unsigned long long*)&lds_h[lrow * LDH + c4 * 4] = p.u;
      }
    }
    if (tid < MROWS) {
      int r = row0 + tid;
      int nid = (r < Bn) ? n_id[r] : 0;
      lds_cl[tid] = vn_index[2 * nid + 1];
    }
  }
  __syncthreads();

  int mloc  = lane & 15;
  int khalf = lane >> 4;

  v8f acc[12];
  {
    v8f z = {0.f,0.f,0.f,0.f,0.f,0.f,0.f,0.f};
    for (int i = 0; i < 12; ++i) acc[i] = z;
  }

  // k outer: A-frags (LDS) reused across 4 N-tiles, B-frags (global) across 3 M-tiles
  for (int k0 = 0; k0 < DIM; k0 += 32) {
    ABFrag a[3];
    for (int m = 0; m < 3; ++m) {
      const _Float16* ap = &lds_h[(m * 16 + mloc) * LDH + k0 + khalf * 8];
      a[m].h[0] = *(const half8*)(ap);
      a[m].h[1] = *(const half8*)(ap + 16);
    }
    for (int j = 0; j < 4; ++j) {
      int ncol = (wave + j * 8) * 16 + mloc;
      ABFrag b;
      const _Float16* bp = Wk16 + (size_t)ncol * DIM + k0 + khalf * 8;
      b.h[0] = *(const half8*)(bp);
      b.h[1] = *(const half8*)(bp + 16);
      for (int m = 0; m < 3; ++m)
        acc[m * 4 + j] = __builtin_amdgcn_wmma_f32_16x16x32_f16(
            false, a[m].v, false, b.v, (short)0, acc[m * 4 + j], false, false);
    }
  }

  // epilogue: tanh(kp + qp[cl] + bias) * Ws, per-lane partial over its columns
  for (int m = 0; m < 3; ++m) {
    float vsum[8];
    for (int r = 0; r < 8; ++r) vsum[r] = 0.f;
    for (int j = 0; j < 4; ++j) {
      int ncol = (wave + j * 8) * 16 + mloc;
      float abv = attn_bias[ncol];
      float wsv = Ws[ncol];
      for (int r = 0; r < 8; ++r) {
        int M = m * 16 + r + khalf * 8;
        int cl = lds_cl[M];
        float v = tanhf(acc[m * 4 + j][r] + qp[(size_t)cl * DIM + ncol] + abv) * wsv;
        vsum[r] += v;
      }
    }
    // butterfly-reduce the 16 columns inside each half-wave
    for (int r = 0; r < 8; ++r) {
      float s = vsum[r];
      s += __shfl_xor(s, 1, 16);
      s += __shfl_xor(s, 2, 16);
      s += __shfl_xor(s, 4, 16);
      s += __shfl_xor(s, 8, 16);
      if (mloc == 0) lds_partial[wave * MROWS + m * 16 + r + khalf * 8] = s;
    }
  }
  __syncthreads();

  if (tid < MROWS) {
    float s = 0.f;
    for (int w = 0; w < 8; ++w) s += lds_partial[w * MROWS + tid];
    s += bs[0];
    int grow = row0 + tid;
    if (grow < Bn) {
      score[grow] = s;
      unsigned int bits = __float_as_uint(s);
      unsigned int key = (bits & 0x80000000u) ? ~bits : (bits | 0x80000000u);
      atomicMax(&smax_enc[lds_cl[tid]], key);   // order-preserving f32 max
    }
  }
}

// ----------------------------------------------------------- softmax pieces
__global__ void exp_kernel(const float* __restrict__ score,
                           const unsigned int* __restrict__ smax_enc,
                           const int* __restrict__ vn_index,
                           const int* __restrict__ n_id,
                           float* __restrict__ e, float* __restrict__ denom,
                           int Bn) {
  int i = blockIdx.x * blockDim.x + threadIdx.x;
  if (i >= Bn) return;
  int cl = vn_index[2 * n_id[i] + 1];
  unsigned int key = smax_enc[cl];
  unsigned int bits = (key & 0x80000000u) ? (key & 0x7FFFFFFFu) : ~key;
  float m = __uint_as_float(bits);
  float ev = expf(score[i] - m);
  e[i] = ev;
  atomicAdd(&denom[cl], ev);
}

// ctx[cl] += (e/denom[cl]) * h[i]; ctx is 2 MB -> L2-resident f32 atomics
__global__ __launch_bounds__(128) void ctx_kernel(
    const float* __restrict__ h, const float* __restrict__ e,
    const float* __restrict__ denom, const int* __restrict__ vn_index,
    const int* __restrict__ n_id, float* __restrict__ ctx, int Bn) {
  int i = blockIdx.x;
  if (i >= Bn) return;
  int cl = vn_index[2 * n_id[i] + 1];
  float attn = e[i] / denom[cl];
  int t = threadIdx.x;                      // 128 threads * float4 = 512
  float4 v = ((const float4*)(h + (size_t)i * DIM))[t];
  float* dst = ctx + (size_t)cl * DIM + 4 * t;
  atomicAdd(dst + 0, attn * v.x);
  atomicAdd(dst + 1, attn * v.y);
  atomicAdd(dst + 2, attn * v.z);
  atomicAdd(dst + 3, attn * v.w);
}

// ---------------------------------------------------------------------------
extern "C" void kernel_launch(void* const* d_in, const int* in_sizes, int n_in,
                              void* d_out, int out_size, void* d_ws, size_t ws_size,
                              hipStream_t stream) {
  const float* h   = (const float*)d_in[0];
  const float* g   = (const float*)d_in[1];
  const int*   vn  = (const int*)  d_in[2];
  const int*   nid = (const int*)  d_in[3];
  const float* Wq  = (const float*)d_in[4];
  const float* Wk  = (const float*)d_in[5];
  const float* ab  = (const float*)d_in[6];
  const float* Ws  = (const float*)d_in[7];
  const float* bs  = (const float*)d_in[8];
  const float* W   = (const float*)d_in[9];
  const float* bb  = (const float*)d_in[10];
  (void)n_in; (void)out_size; (void)ws_size;

  const int D  = DIM;
  const int Bn = in_sizes[0] / D;   // 100000
  const int C  = in_sizes[1] / D;   // 1000

  char* base = (char*)d_ws;
  size_t off = 0;
  auto take = [&](size_t bytes) -> char* {
    char* p = base + off;
    off += (bytes + 255) & ~(size_t)255;
    return p;
  };
  _Float16* Wq16  = (_Float16*)take((size_t)D * D * 2);
  _Float16* Wk16  = (_Float16*)take((size_t)D * D * 2);
  _Float16* W16   = (_Float16*)take((size_t)D * D * 2);
  _Float16* g16   = (_Float16*)take((size_t)C * D * 2);
  _Float16* ctx16 = (_Float16*)take((size_t)C * D * 2);
  float*        qp   = (float*)take((size_t)C * D * 4);
  unsigned int* smax = (unsigned int*)take((size_t)(C * D + 2 * C) * 4);
  float*        denom = (float*)(smax + C);
  float*        ctx   = (float*)(smax + 2 * C);
  float*        score = (float*)take((size_t)Bn * 4);
  float*        ev    = (float*)take((size_t)Bn * 4);

  const int thr = 256;
  // 1. precision conversions
  cvt_f32_f16_kernel<<<(D * D / 4 + thr - 1) / thr, thr, 0, stream>>>(Wq, Wq16, D * D / 4);
  cvt_f32_f16_kernel<<<(D * D / 4 + thr - 1) / thr, thr, 0, stream>>>(Wk, Wk16, D * D / 4);
  cvt_f32_f16_kernel<<<(D * D / 4 + thr - 1) / thr, thr, 0, stream>>>(W,  W16,  D * D / 4);
  cvt_f32_f16_kernel<<<(C * D / 4 + thr - 1) / thr, thr, 0, stream>>>(g,  g16,  C * D / 4);
  // 2. zero smax/denom/ctx (re-zeroed every call: graph-replay safe)
  int zn = C * D + 2 * C;
  zero_kernel<<<(zn + thr - 1) / thr, thr, 0, stream>>>(smax, zn);
  // 3. qp = g @ Wq^T
  dim3 gq((C + 15) / 16, D / 128);
  wmma_gemm_nt_kernel<<<gq, 256, 0, stream>>>(g16, Wq16, nullptr, qp, C, D, 0);
  // 4. fused kp/score + segment max
  score_kernel<<<(Bn + MROWS - 1) / MROWS, 256, 0, stream>>>(h, Wk16, qp, vn, nid,
                                                             ab, Ws, bs, score, smax, Bn);
  // 5. exp + segment sum
  exp_kernel<<<(Bn + 255) / 256, 256, 0, stream>>>(score, smax, vn, nid, ev, denom, Bn);
  // 6. attention-weighted segment sum of h
  ctx_kernel<<<Bn, 128, 0, stream>>>(h, ev, denom, vn, nid, ctx, Bn);
  // 7-8. final projection + tanh
  cvt_f32_f16_kernel<<<(C * D / 4 + thr - 1) / thr, thr, 0, stream>>>(ctx, ctx16, C * D / 4);
  dim3 go((C + 15) / 16, D / 128);
  wmma_gemm_nt_kernel<<<go, 256, 0, stream>>>(ctx16, W16, bb, (float*)d_out, C, D, 1);
}
